// RNN_57458072486091
// MI455X (gfx1250) — matrix operand order repaired and checked by reading, not compiled, not run on hardware
//
#include <hip/hip_runtime.h>
#include <stdint.h>

// ---------------- problem constants ----------------
#define B_  32
#define T_  1024
#define I_  512
#define H_  1024
#define O_  512

// ---------------- workspace layout (bytes) ----------------
static constexpr size_t OFF_XH   = 0;                                  // f32 [T][B][H]
static constexpr size_t OFF_HS   = OFF_XH  + (size_t)T_*B_*H_*4;       // bf16 [T][B][H]
static constexpr size_t OFF_XB   = OFF_HS  + (size_t)T_*B_*H_*2;       // bf16 [T*B][I]
static constexpr size_t OFF_PWXH = OFF_XB  + (size_t)T_*B_*I_*2;       // packed W_xh
static constexpr size_t OFF_PWHH = OFF_PWXH + (size_t)I_*H_*2;         // packed W_hh
static constexpr size_t OFF_PWHY = OFF_PWHH + (size_t)H_*H_*2;         // packed W_hy
static constexpr size_t OFF_BAR  = OFF_PWHY + (size_t)H_*O_*2;         // barrier

// ---------------- WMMA types ----------------
typedef __attribute__((ext_vector_type(16))) __bf16          v16bf;
typedef __attribute__((ext_vector_type(8)))  float           v8f;
typedef __attribute__((ext_vector_type(16))) unsigned short  v16u;
typedef __attribute__((ext_vector_type(8)))  unsigned short  v8u;

static __device__ __forceinline__ unsigned short f2bf(float f) {
    unsigned u = __builtin_bit_cast(unsigned, f);
    u += 0x7FFFu + ((u >> 16) & 1u);       // round-to-nearest-even
    return (unsigned short)(u >> 16);
}

static __device__ __forceinline__ v8f wmma_bf16(v16u a, v16u b, v8f c) {
    return __builtin_amdgcn_wmma_f32_16x16x32_bf16(
        false, __builtin_bit_cast(v16bf, a),
        false, __builtin_bit_cast(v16bf, b),
        (short)0, c, false, false);
}

// A fragment (16x32 bf16, row-major source, stride in elements).
static __device__ __forceinline__ v16u load_a(const unsigned short* __restrict__ base,
                                              int stride, int row0, int k0, int lane) {
    const unsigned short* p = base + (size_t)(row0 + (lane & 15)) * stride
                                   + k0 + ((lane >> 4) << 3);
    v8u lo = *(const v8u*)(p);
    v8u hi = *(const v8u*)(p + 16);
    return __builtin_shufflevector(lo, hi, 0,1,2,3,4,5,6,7,8,9,10,11,12,13,14,15);
}

// B fragment from packed buffer: 512 contiguous bf16 per fragment (lane*16+j).
static __device__ __forceinline__ v16u load_b(const unsigned short* __restrict__ frag,
                                              int lane) {
    const unsigned short* p = frag + lane * 16;
    v8u lo = *(const v8u*)(p);
    v8u hi = *(const v8u*)(p + 8);
    return __builtin_shufflevector(lo, hi, 0,1,2,3,4,5,6,7,8,9,10,11,12,13,14,15);
}

// ---------------- prep kernels ----------------
__global__ void k_zero_bar(unsigned int* bar) {
    if (threadIdx.x == 0 && blockIdx.x == 0) bar[0] = 0u;
}

// Pack W [K][N] f32 row-major -> WMMA B fragments (bf16).
__global__ void k_pack_w(const float* __restrict__ src, unsigned short* __restrict__ dst,
                         int K, int N) {
    int p = blockIdx.x * blockDim.x + threadIdx.x;
    if (p >= K * N) return;
    int f    = p >> 9;
    int r    = p & 511;
    int lane = r >> 4;
    int j    = r & 15;
    int ktiles = K >> 5;
    int nt = f / ktiles;
    int kt = f - nt * ktiles;
    int n  = nt * 16 + (lane & 15);
    int k  = kt * 32 + ((lane >> 4) << 3) + (j & 7) + ((j >> 3) << 4);
    dst[p] = f2bf(src[(size_t)k * N + n]);
}

// x [B][T][I] f32 -> xb [R=t*B+b][I] bf16
__global__ void k_pack_x(const float* __restrict__ x, unsigned short* __restrict__ xb) {
    int q = blockIdx.x * blockDim.x + threadIdx.x;
    int R = q >> 9;
    int i = q & 511;
    int t = R >> 5;
    int b = R & 31;
    xb[q] = f2bf(x[((size_t)b * T_ + t) * I_ + i]);
}

// ---------------- GEMM1: xh[t*B+b][H] = xb @ W_xh + b_h ----------------
// 2x2 (32x32) tile block per wave: M-pairs=1024, N-pairs=32 -> 32768 waves
__global__ void __launch_bounds__(256) k_gemm1(const unsigned short* __restrict__ xb,
                                               const unsigned short* __restrict__ pW,
                                               const float* __restrict__ bh,
                                               float* __restrict__ xh) {
    int wv   = (blockIdx.x * blockDim.x + threadIdx.x) >> 5;
    int lane = threadIdx.x & 31;
    int tmp = wv >> 5;                 // 0..1023  (M pair)
    int tnp = wv & 31;                 // 0..31    (N pair)
    int R0 = tmp << 5, n0 = tnp << 5;
    v8f c00 = {}, c01 = {}, c10 = {}, c11 = {};
    const unsigned short* bp0 = pW + (size_t)(2 * tnp)     * (I_ / 32) * 512;
    const unsigned short* bp1 = pW + (size_t)(2 * tnp + 1) * (I_ / 32) * 512;
    #pragma unroll 2
    for (int kt = 0; kt < I_ / 32; ++kt) {
        v16u a0 = load_a(xb, I_, R0,      kt * 32, lane);
        v16u a1 = load_a(xb, I_, R0 + 16, kt * 32, lane);
        v16u b0 = load_b(bp0 + (size_t)kt * 512, lane);
        v16u b1 = load_b(bp1 + (size_t)kt * 512, lane);
        c00 = wmma_bf16(a0, b0, c00);
        c01 = wmma_bf16(a0, b1, c01);
        c10 = wmma_bf16(a1, b0, c10);
        c11 = wmma_bf16(a1, b1, c11);
    }
    int col0 = n0 + (lane & 15), col1 = col0 + 16;
    float bia0 = bh[col0], bia1 = bh[col1];
    int rb0 = R0 + ((lane >> 4) << 3), rb1 = rb0 + 16;
    #pragma unroll
    for (int r = 0; r < 8; ++r) {
        xh[(size_t)(rb0 + r) * H_ + col0] = c00[r] + bia0;
        xh[(size_t)(rb0 + r) * H_ + col1] = c01[r] + bia1;
        xh[(size_t)(rb1 + r) * H_ + col0] = c10[r] + bia0;
        xh[(size_t)(rb1 + r) * H_ + col1] = c11[r] + bia1;
    }
}

// ---------------- persistent recurrence ----------------
// grid = 8 WGs x 256 threads (8 waves). WG w owns columns [128w, 128w+128);
// its 1024x128 W_hh slice (256 KB bf16) is async-loaded into LDS once and
// reused for all 1024 timesteps (CDNA5: 320 KB LDS per WGP).
// wave wid: m0 = 16*(wid&1); N-tile pair npair = wid>>1 (0..3):
//   one A fragment feeds two WMMAs (two adjacent N-tiles).
__global__ void __launch_bounds__(256, 1) k_rnn(const float* __restrict__ xh,
                                                const unsigned short* __restrict__ pWhh,
                                                unsigned short* __restrict__ hs,
                                                unsigned int* __restrict__ bar) {
    extern __shared__ unsigned short sW[];   // 131072 ushorts = 256 KB
    const int w    = blockIdx.x;             // 0..7
    const int tid  = threadIdx.x;
    const int wid  = tid >> 5;
    const int lane = tid & 31;

    // ---- async preload of this WG's W_hh slice: 262144 bytes ----
    // fragments [256w, 256w+256) are contiguous in the packed buffer.
    {
        const char* gsrc = (const char*)(pWhh + (size_t)w * 131072);
        // no static __shared__ in this kernel -> dynamic LDS base offset is 0
        for (int i = 0; i < 64; ++i) {
            unsigned chunk   = (unsigned)tid + (unsigned)i * 256u;   // 16-byte chunks
            unsigned lds_off = chunk * 16u;
            unsigned long long gaddr = (unsigned long long)(gsrc) + (unsigned long long)lds_off;
            asm volatile("global_load_async_to_lds_b128 %0, %1, off"
                         :: "v"(lds_off), "v"(gaddr) : "memory");
        }
        asm volatile("s_wait_asynccnt 0" ::: "memory");
    }
    __syncthreads();

    const int m0    = (wid & 1) << 4;             // 0 or 16
    const int npair = wid >> 1;                   // 0..3
    const int ln0   = npair << 1;                 // local N-tile 0..7 (pairs)
    const int col0  = (w << 7) + (ln0 << 4) + (lane & 15);
    const int col1  = col0 + 16;
    const int rb    = m0 + ((lane >> 4) << 3);

    for (int t = 0; t < T_; ++t) {
        v8f c0 = {}, c1 = {};
        if (t > 0) {
            const unsigned short* Abase = hs + (size_t)(t - 1) * B_ * H_;
            #pragma unroll 4
            for (int kt = 0; kt < H_ / 32; ++kt) {
                v16u a = load_a(Abase, H_, m0, kt * 32, lane);
                const unsigned short* q0 = sW + ((size_t)((ln0    ) * 32 + kt) * 512) + lane * 16;
                const unsigned short* q1 = sW + ((size_t)((ln0 + 1) * 32 + kt) * 512) + lane * 16;
                v8u l0 = *(const v8u*)(q0);
                v8u h0 = *(const v8u*)(q0 + 8);
                v8u l1 = *(const v8u*)(q1);
                v8u h1 = *(const v8u*)(q1 + 8);
                v16u b0 = __builtin_shufflevector(l0, h0, 0,1,2,3,4,5,6,7,8,9,10,11,12,13,14,15);
                v16u b1 = __builtin_shufflevector(l1, h1, 0,1,2,3,4,5,6,7,8,9,10,11,12,13,14,15);
                c0 = wmma_bf16(a, b0, c0);
                c1 = wmma_bf16(a, b1, c1);
            }
        }
        // epilogue: h = tanh(acc + xh_t) for both N-tiles, store bf16
        #pragma unroll
        for (int r = 0; r < 8; ++r) {
            int m = rb + r;
            size_t row = (size_t)t * B_ + m;
            float v0 = tanhf(c0[r] + xh[row * H_ + col0]);
            float v1 = tanhf(c1[r] + xh[row * H_ + col1]);
            hs[row * H_ + col0] = f2bf(v0);
            hs[row * H_ + col1] = f2bf(v1);
        }
        if (t + 1 < T_)
            __builtin_prefetch(&xh[((size_t)(t + 1) * B_ + rb) * H_ + col0], 0, 0);
        // device-wide step barrier (8 workgroups)
        __syncthreads();
        if (tid == 0) {
            __threadfence();
            __hip_atomic_fetch_add(bar, 1u, __ATOMIC_RELEASE, __HIP_MEMORY_SCOPE_AGENT);
            const unsigned tgt = 8u * (unsigned)(t + 1);
            while (__hip_atomic_load(bar, __ATOMIC_ACQUIRE, __HIP_MEMORY_SCOPE_AGENT) < tgt)
                __builtin_amdgcn_s_sleep(2);
        }
        __syncthreads();
    }
}

// ---------------- GEMM3: y[b][t][O] = hs @ W_hy + b_y ----------------
// 2x2 tile block per wave: M-pairs=1024, N-pairs=16 -> 16384 waves
__global__ void __launch_bounds__(256) k_gemm3(const unsigned short* __restrict__ hs,
                                               const unsigned short* __restrict__ pW,
                                               const float* __restrict__ by,
                                               float* __restrict__ y) {
    int wv   = (blockIdx.x * blockDim.x + threadIdx.x) >> 5;
    int lane = threadIdx.x & 31;
    int tmp = wv >> 4;                 // 0..1023
    int tnp = wv & 15;                 // 0..15
    int R0 = tmp << 5, n0 = tnp << 5;
    v8f c00 = {}, c01 = {}, c10 = {}, c11 = {};
    const unsigned short* bp0 = pW + (size_t)(2 * tnp)     * (H_ / 32) * 512;
    const unsigned short* bp1 = pW + (size_t)(2 * tnp + 1) * (H_ / 32) * 512;
    #pragma unroll 2
    for (int kt = 0; kt < H_ / 32; ++kt) {
        v16u a0 = load_a(hs, H_, R0,      kt * 32, lane);
        v16u a1 = load_a(hs, H_, R0 + 16, kt * 32, lane);
        v16u b0 = load_b(bp0 + (size_t)kt * 512, lane);
        v16u b1 = load_b(bp1 + (size_t)kt * 512, lane);
        c00 = wmma_bf16(a0, b0, c00);
        c01 = wmma_bf16(a0, b1, c01);
        c10 = wmma_bf16(a1, b0, c10);
        c11 = wmma_bf16(a1, b1, c11);
    }
    int o0 = n0 + (lane & 15), o1 = o0 + 16;
    float bia0 = by[o0], bia1 = by[o1];
    int rb0 = R0 + ((lane >> 4) << 3), rb1 = rb0 + 16;
    #pragma unroll
    for (int r = 0; r < 8; ++r) {
        int Ra = rb0 + r, Rb = rb1 + r;
        int ta = Ra >> 5, ba = Ra & 31;
        int tb = Rb >> 5, bb = Rb & 31;
        y[((size_t)ba * T_ + ta) * O_ + o0] = c00[r] + bia0;
        y[((size_t)ba * T_ + ta) * O_ + o1] = c01[r] + bia1;
        y[((size_t)bb * T_ + tb) * O_ + o0] = c10[r] + bia0;
        y[((size_t)bb * T_ + tb) * O_ + o1] = c11[r] + bia1;
    }
}

// ---------------- host launcher ----------------
extern "C" void kernel_launch(void* const* d_in, const int* in_sizes, int n_in,
                              void* d_out, int out_size, void* d_ws, size_t ws_size,
                              hipStream_t stream) {
    const float* x    = (const float*)d_in[0];
    const float* Wxh  = (const float*)d_in[1];
    const float* Whh  = (const float*)d_in[2];
    const float* Why  = (const float*)d_in[3];
    const float* bh   = (const float*)d_in[4];
    const float* by   = (const float*)d_in[5];
    float*       y    = (float*)d_out;

    char* ws = (char*)d_ws;
    float*          xh   = (float*)(ws + OFF_XH);
    unsigned short* hs   = (unsigned short*)(ws + OFF_HS);
    unsigned short* xb   = (unsigned short*)(ws + OFF_XB);
    unsigned short* pWxh = (unsigned short*)(ws + OFF_PWXH);
    unsigned short* pWhh = (unsigned short*)(ws + OFF_PWHH);
    unsigned short* pWhy = (unsigned short*)(ws + OFF_PWHY);
    unsigned int*   bar  = (unsigned int*)(ws + OFF_BAR);

    k_zero_bar<<<1, 64, 0, stream>>>(bar);
    k_pack_w<<<(I_ * H_ + 255) / 256, 256, 0, stream>>>(Wxh, pWxh, I_, H_);
    k_pack_w<<<(H_ * H_ + 255) / 256, 256, 0, stream>>>(Whh, pWhh, H_, H_);
    k_pack_w<<<(H_ * O_ + 255) / 256, 256, 0, stream>>>(Why, pWhy, H_, O_);
    k_pack_x<<<(T_ * B_ * I_) / 256, 256, 0, stream>>>(x, xb);

    // GEMM1: 32768 waves (2x2 tiles each), 8 waves per WG
    k_gemm1<<<32768 / 8, 256, 0, stream>>>(xb, pWxh, bh, xh);

    // persistent recurrence: 8 WGs, 256 KB dynamic LDS each
    k_rnn<<<8, 256, 262144, stream>>>(xh, pWhh, hs, bar);

    // GEMM3: 16384 waves (2x2 tiles each), 8 waves per WG
    k_gemm3<<<16384 / 8, 256, 0, stream>>>(hs, pWhy, by, y);
}
